// GraphCNN_26817775796449
// MI455X (gfx1250) — matrix operand order repaired and checked by reading, not compiled, not run on hardware
//
#include <hip/hip_runtime.h>

#define G_ 32
#define N_ 1024
#define F_ 128
#define H_ 128
#define L_ 5
#define O_ 16

typedef __attribute__((ext_vector_type(16))) _Float16 v16h;
typedef __attribute__((ext_vector_type(8)))  _Float16 v8h;
typedef __attribute__((ext_vector_type(8)))  float    v8f;
typedef __attribute__((ext_vector_type(4)))  float    v4f;

union AccU { v8f v; float f[8]; };   // constant-index access only (no spills)

__device__ __forceinline__ v8f wmma_f16(v16h a, v16h b, v8f c) {
  // D = A(16x32,f16) * B(32x16,f16) + C(f32)
  return __builtin_amdgcn_wmma_f32_16x16x32_f16(false, a, false, b, (short)0, c, false, false);
}

__device__ __forceinline__ v16h cat16(v8h lo, v8h hi) {
  return __builtin_shufflevector(lo, hi, 0,1,2,3,4,5,6,7,8,9,10,11,12,13,14,15);
}

// ---------------------------------------------------------------------------
// adj fp32 [G,N,N] -> f16 with identity baked in (adj_sl = adj + I).
// One thread = 8 consecutive elements of one row.
// ---------------------------------------------------------------------------
__global__ void adj_convert_kernel(const float* __restrict__ adj,
                                   _Float16* __restrict__ adjh) {
  size_t chunk = (size_t)blockIdx.x * blockDim.x + threadIdx.x;
  size_t base = chunk * 8;                 // G*N*N total elements
  int r  = (int)((base >> 10) & (N_ - 1)); // row within graph
  int c0 = (int)(base & (N_ - 1));         // col of first element
  const float* src = adj + base;
  v8h pk;
#pragma unroll
  for (int i = 0; i < 8; ++i) {
    float x = src[i] + ((c0 + i == r) ? 1.0f : 0.0f);
    pk[i] = (_Float16)x;
  }
  *(v8h*)(adjh + base) = pk;
}

// ---------------------------------------------------------------------------
// Prep: transpose+convert MLP weights to f16 (Wt[l][j][k] = W[l][k][j]),
// zero the per-layer graph sums.
// ---------------------------------------------------------------------------
__global__ void prep_kernel(const float* __restrict__ W1, const float* __restrict__ W2,
                            _Float16* __restrict__ W1t, _Float16* __restrict__ W2t,
                            float* __restrict__ layer_sum) {
  int i = blockIdx.x * blockDim.x + threadIdx.x;
  const int tw = L_ * H_ * H_;
  if (i < tw) {
    int l = i / (H_ * H_);
    int j = (i / H_) % H_;
    int k = i % H_;
    int src = l * H_ * H_ + k * H_ + j;
    W1t[i] = (_Float16)W1[src];
    W2t[i] = (_Float16)W2[src];
  }
  if (i < (L_ + 1) * G_ * H_) layer_sum[i] = 0.0f;
}

// ---------------------------------------------------------------------------
// h0: node_features [G,N,F] fp32 -> ht [G,H,N] f16 (transposed), plus
// layer-0 sum-pool into layer_sum[0][g][c].
// ---------------------------------------------------------------------------
__global__ void init_h_kernel(const float* __restrict__ nf, _Float16* __restrict__ ht,
                              float* __restrict__ lsum0) {
  int g = blockIdx.x;
  int c = threadIdx.x;     // 128 threads
  const float* src = nf + (size_t)g * N_ * F_ + c;
  _Float16* dst = ht + ((size_t)g * H_ + c) * N_;
  float s = 0.0f;
  for (int n = 0; n < N_; n += 8) {
    v8h pk;
#pragma unroll
    for (int i = 0; i < 8; ++i) {
      float x = src[(size_t)(n + i) * F_];
      s += x;
      pk[i] = (_Float16)x;
    }
    *(v8h*)(dst + n) = pk;
  }
  lsum0[(size_t)g * H_ + c] = s;
}

// ---------------------------------------------------------------------------
// Fused GIN layer. Block: 256 thr = 8 waves, covers 128 nodes x H=128 of one
// graph. Grid: G*8.
//   ADJ16=true : A-fragments from pre-converted f16 adj (+I baked in).
//   ADJ16=false: A-fragments converted in-register from fp32 adj; +I applied
//                analytically afterwards (pooled += h).
// ---------------------------------------------------------------------------
template<bool ADJ16>
__global__ __launch_bounds__(256) void gin_layer_kernel(
    const float* __restrict__ adjf,     // [G,N,N] fp32 (no self loops)
    const _Float16* __restrict__ adjh,  // [G,N,N] f16 (+I)
    const _Float16* __restrict__ htin,  // [G,H,N] f16 transposed
    _Float16* __restrict__ htout,       // [G,H,N] f16 transposed
    const _Float16* __restrict__ W1t,   // [H,H] f16, W1^T
    const _Float16* __restrict__ W2t,   // [H,H] f16, W2^T
    const float* __restrict__ b1,
    const float* __restrict__ b2,
    const float* __restrict__ lng,
    const float* __restrict__ lnb,
    float* __restrict__ lsum)           // [G,H] this layer's readout slot
{
  __shared__ __align__(16) _Float16 lsmat[128 * 136];
  __shared__ float colsum[128];

  const int tid   = threadIdx.x;
  const int g     = blockIdx.x >> 3;
  const int node0 = (blockIdx.x & 7) * 128;
  const int wid   = tid >> 5;
  const int lane  = tid & 31;
  const int lrow  = lane & 15;
  const int hi    = lane >> 4;
  const int kA    = hi * 16;  // A-frag K base (per half-wave)
  const int kB    = hi * 16;  // B-frag K base (per half-wave)
  const int wrow0 = wid * 16; // row base inside the 128-row tile
  const int m     = node0 + wrow0 + lrow; // this lane's A row (global node id)

  if (tid < 128) colsum[tid] = 0.0f;

  // ---- Stage 1: pooled = adj_sl @ h  via f16 WMMA, fp32 accumulate ----
  AccU acc[8];
#pragma unroll
  for (int t = 0; t < 8; ++t) { v8f zz = {}; acc[t].v = zz; }

  const _Float16* hb = htin + (size_t)g * H_ * N_;

  if constexpr (ADJ16) {
    const _Float16* arow = adjh + ((size_t)g * N_ + m) * N_;
#pragma unroll 2
    for (int k0 = 0; k0 < N_; k0 += 32) {
      v8h lo = *(const v8h*)(arow + k0 + kA);
      v8h hh = *(const v8h*)(arow + k0 + kA + 8);
      v16h a = cat16(lo, hh);
#pragma unroll
      for (int t = 0; t < 8; ++t) {
        v16h b = *(const v16h*)(hb + (size_t)(16 * t + lrow) * N_ + (k0 + kB));
        acc[t].v = wmma_f16(a, b, acc[t].v);
      }
    }
  } else {
    const float* arow = adjf + ((size_t)g * N_ + m) * N_;
#pragma unroll 2
    for (int k0 = 0; k0 < N_; k0 += 32) {
      const float* ap = arow + k0 + kA;
      v4f x0 = *(const v4f*)(ap);
      v4f x1 = *(const v4f*)(ap + 4);
      v4f x2 = *(const v4f*)(ap + 8);
      v4f x3 = *(const v4f*)(ap + 12);
      v16h a;
#pragma unroll
      for (int i = 0; i < 4; ++i) {
        a[i]      = (_Float16)x0[i];
        a[i + 4]  = (_Float16)x1[i];
        a[i + 8]  = (_Float16)x2[i];
        a[i + 12] = (_Float16)x3[i];
      }
#pragma unroll
      for (int t = 0; t < 8; ++t) {
        v16h b = *(const v16h*)(hb + (size_t)(16 * t + lrow) * N_ + (k0 + kB));
        acc[t].v = wmma_f16(a, b, acc[t].v);
      }
    }
    // +I analytically: pooled[r][c] += h[r][c]; rows v..v+7 are contiguous in ht
#pragma unroll
    for (int t = 0; t < 8; ++t) {
      const _Float16* hp = hb + (size_t)(16 * t + lrow) * N_ + node0 + wrow0 + 8 * hi;
      v8h hv = *(const v8h*)hp;
#pragma unroll
      for (int v = 0; v < 8; ++v) acc[t].f[v] += (float)hv[v];
    }
  }

  // pooled (f16) -> LDS row-major [row][col], stride 136 halfs (bank-safe)
#pragma unroll
  for (int v = 0; v < 8; ++v)
#pragma unroll
    for (int t = 0; t < 8; ++t)
      lsmat[(wrow0 + v + 8 * hi) * 136 + 16 * t + lrow] = (_Float16)acc[t].f[v];
  __syncthreads();

  // per-lane column params (cols c = 16t + lrow)
  float b1c[8], b2c[8], gc[8], bc[8];
#pragma unroll
  for (int t = 0; t < 8; ++t) {
    int c = 16 * t + lrow;
    b1c[t] = b1[c]; b2c[t] = b2[c]; gc[t] = lng[c]; bc[t] = lnb[c];
  }

  // ---- Stage 2a: z1 = relu(pooled @ W1 + b1) ----
  AccU z1[8];
#pragma unroll
  for (int t = 0; t < 8; ++t) { v8f zz = {}; z1[t].v = zz; }
  const _Float16* arow_l = &lsmat[(wrow0 + lrow) * 136];
#pragma unroll
  for (int ks = 0; ks < 4; ++ks) {
    v16h a = cat16(*(const v8h*)(arow_l + 32 * ks + kA),
                   *(const v8h*)(arow_l + 32 * ks + kA + 8));
#pragma unroll
    for (int t = 0; t < 8; ++t) {
      v16h b = *(const v16h*)(W1t + (size_t)(16 * t + lrow) * H_ + 32 * ks + kB);
      z1[t].v = wmma_f16(a, b, z1[t].v);
    }
  }
  __syncthreads();  // everyone done reading pooled
#pragma unroll
  for (int t = 0; t < 8; ++t)
#pragma unroll
    for (int v = 0; v < 8; ++v) {
      float z = z1[t].f[v] + b1c[t];
      lsmat[(wrow0 + v + 8 * hi) * 136 + 16 * t + lrow] = (_Float16)(z > 0.0f ? z : 0.0f);
    }
  __syncthreads();

  // ---- Stage 2b: z2 = z1 @ W2 + b2 ----
  AccU z2[8];
#pragma unroll
  for (int t = 0; t < 8; ++t) { v8f zz = {}; z2[t].v = zz; }
#pragma unroll
  for (int ks = 0; ks < 4; ++ks) {
    v16h a = cat16(*(const v8h*)(arow_l + 32 * ks + kA),
                   *(const v8h*)(arow_l + 32 * ks + kA + 8));
#pragma unroll
    for (int t = 0; t < 8; ++t) {
      v16h b = *(const v16h*)(W2t + (size_t)(16 * t + lrow) * H_ + 32 * ks + kB);
      z2[t].v = wmma_f16(a, b, z2[t].v);
    }
  }
  __syncthreads();  // everyone done reading z1; lsmat reusable

  // ---- Stage 3: LayerNorm over H (each row lives in one 16-lane half) ----
#pragma unroll
  for (int v = 0; v < 8; ++v) {
    float s = 0.0f, ss = 0.0f;
#pragma unroll
    for (int t = 0; t < 8; ++t) {
      float x = z2[t].f[v] + b2c[t];
      z2[t].f[v] = x;
      s += x; ss += x * x;
    }
#pragma unroll
    for (int msk = 1; msk <= 8; msk <<= 1) {
      s  += __shfl_xor(s,  msk, 32);
      ss += __shfl_xor(ss, msk, 32);
    }
    float mu   = s * (1.0f / 128.0f);
    float var  = ss * (1.0f / 128.0f) - mu * mu;
    float rstd = rsqrtf(var + 1e-5f);
#pragma unroll
    for (int t = 0; t < 8; ++t) {
      float y = (z2[t].f[v] - mu) * rstd * gc[t] + bc[t];
      z2[t].f[v] = y > 0.0f ? y : 0.0f;
    }
  }

  // ---- Stage 4: stash h in LDS, column sums, coalesced transposed writeback
#pragma unroll
  for (int t = 0; t < 8; ++t) {
    float cs = 0.0f;
#pragma unroll
    for (int v = 0; v < 8; ++v) {
      float hval = z2[t].f[v];
      cs += hval;
      lsmat[(wrow0 + v + 8 * hi) * 136 + 16 * t + lrow] = (_Float16)hval;
    }
    atomicAdd(&colsum[16 * t + lrow], cs);
  }
  __syncthreads();

  {
    int c = tid & 127, seg = tid >> 7;
    _Float16* outp = htout + ((size_t)g * H_ + c) * N_ + node0 + seg * 64;
#pragma unroll
    for (int j = 0; j < 64; j += 8) {
      v8h pk;
#pragma unroll
      for (int i = 0; i < 8; ++i)
        pk[i] = lsmat[(seg * 64 + j + i) * 136 + c];
      *(v8h*)(outp + j) = pk;
    }
  }
  if (tid < 128) atomicAdd(&lsum[(size_t)g * H_ + tid], colsum[tid]);
}

// ---------------------------------------------------------------------------
// Readout: score[g][o] = sum_l ( layer_sum[l][g] . pred_W[l][:,o] + pred_b[l][o] )
// ---------------------------------------------------------------------------
__global__ void readout_kernel(const float* __restrict__ layer_sum,
                               const float* __restrict__ predW,
                               const float* __restrict__ predb,
                               float* __restrict__ out) {
  int t = threadIdx.x;  // 512 = G*O
  int g = t >> 4, o = t & 15;
  float s = 0.0f;
  for (int l = 0; l <= L_; ++l) {
    const float* ls = layer_sum + ((size_t)l * G_ + g) * H_;
    const float* pw = predW + (size_t)l * H_ * O_ + o;
    float acc = 0.0f;
    for (int h = 0; h < H_; ++h) acc += ls[h] * pw[(size_t)h * O_];
    s += acc + predb[l * O_ + o];
  }
  out[(size_t)g * O_ + o] = s;
}

// ---------------------------------------------------------------------------
extern "C" void kernel_launch(void* const* d_in, const int* in_sizes, int n_in,
                              void* d_out, int out_size, void* d_ws, size_t ws_size,
                              hipStream_t stream) {
  const float* nf  = (const float*)d_in[0];   // [G,N,F]
  const float* adj = (const float*)d_in[1];   // [G,N,N]
  const float* W1  = (const float*)d_in[2];   // [L,H,H]
  const float* b1  = (const float*)d_in[3];   // [L,H]
  const float* W2  = (const float*)d_in[4];   // [L,H,H]
  const float* b2  = (const float*)d_in[5];   // [L,H]
  const float* lng = (const float*)d_in[6];   // [L,H]
  const float* lnb = (const float*)d_in[7];   // [L,H]
  const float* pW  = (const float*)d_in[8];   // [L+1,H,O]
  const float* pb  = (const float*)d_in[9];   // [L+1,O]
  float* out = (float*)d_out;                 // [G,O]

  const size_t ADJ = (size_t)G_ * N_ * N_;    // 33,554,432 halfs (64 MB)
  const size_t HT  = (size_t)G_ * H_ * N_;    // 4,194,304 halfs (8 MB)
  const size_t WT  = (size_t)L_ * H_ * H_;    // 81,920 halfs
  const size_t tail_bytes = (2 * HT + 2 * WT) * sizeof(_Float16)
                          + (size_t)(L_ + 1) * G_ * H_ * sizeof(float);
  const bool use_adj16 = ws_size >= ADJ * sizeof(_Float16) + tail_bytes;

  _Float16* adjh = (_Float16*)d_ws;
  _Float16* ht_a = use_adj16 ? (adjh + ADJ) : (_Float16*)d_ws;
  _Float16* ht_b = ht_a + HT;
  _Float16* W1t  = ht_b + HT;
  _Float16* W2t  = W1t + WT;
  float* layer_sum = (float*)(W2t + WT);      // [(L+1), G, H]

  if (use_adj16)
    adj_convert_kernel<<<(int)(ADJ / 8 / 256), 256, 0, stream>>>(adj, adjh);
  prep_kernel<<<(int)((WT + 255) / 256), 256, 0, stream>>>(W1, W2, W1t, W2t, layer_sum);
  init_h_kernel<<<G_, 128, 0, stream>>>(nf, ht_a, layer_sum);

  _Float16* hin = ht_a;
  _Float16* hout = ht_b;
  for (int l = 0; l < L_; ++l) {
    if (use_adj16)
      gin_layer_kernel<true><<<G_ * 8, 256, 0, stream>>>(
          adj, adjh, hin, hout,
          W1t + (size_t)l * H_ * H_, W2t + (size_t)l * H_ * H_,
          b1 + l * H_, b2 + l * H_, lng + l * H_, lnb + l * H_,
          layer_sum + (size_t)(l + 1) * G_ * H_);
    else
      gin_layer_kernel<false><<<G_ * 8, 256, 0, stream>>>(
          adj, adjh, hin, hout,
          W1t + (size_t)l * H_ * H_, W2t + (size_t)l * H_ * H_,
          b1 + l * H_, b2 + l * H_, lng + l * H_, lnb + l * H_,
          layer_sum + (size_t)(l + 1) * G_ * H_);
    _Float16* tmp = hin; hin = hout; hout = tmp;
  }
  readout_kernel<<<1, 512, 0, stream>>>(layer_sum, pW, pb, out);
}
